// SpanNgramAttentions_45380624450152
// MI455X (gfx1250) — matrix operand order repaired and checked by reading, not compiled, not run on hardware
//
#include <hip/hip_runtime.h>

// Problem constants (from reference)
#define CCH 8        // channels
#define LL  256      // ngram sequence length
#define DD  256      // embedding dim
#define NSP 16384    // S*S span rows
#define CD  2048     // C*D output row width
#define RPB 64       // span rows per block
#define PE  264      // LDS pitch (f16 elems) for emb / embT rows -> 528B stride, bank-friendly
#define PA  264      // LDS pitch (f16 elems) for A tiles

typedef __attribute__((ext_vector_type(16))) _Float16 v16h;
typedef __attribute__((ext_vector_type(8)))  float    v8f;

// LDS layout (bytes):
//   embS   [256][PE] f16 : 135168   @ 0
//   embT   [256][PE] f16 : 135168   @ 135168  (transposed copy: row d, col l)
//   aS     [16][PA]  f16 : 8448     @ 270336  (span tile fragments, pre-scaled 1/16)
//   attS   [16][PA]  f16 : 8448     @ 278784  (unnormalized exp(u)*mask, f16)
//   rowsum [16]      f32 : 64       @ 287232
//   stage  [16][256] f32 : 16384    @ 287296  (async span staging, 16B aligned)
#define SMEM_BYTES 303680

union FragU { v16h v; unsigned u[8]; };

__device__ __forceinline__ void load_fragA(FragU& A, const unsigned* arow, int k, int h) {
    #pragma unroll
    for (int q = 0; q < 8; ++q) {
        int kbA = ((q < 4) ? 2*q : 16 + 2*(q-4)) + 8*h;   // 16-bit A 16x32 VGPR layout
        A.u[q] = arow[(k + kbA) >> 1];
    }
}
__device__ __forceinline__ void load_fragB(FragU& B, const unsigned* brow, int k, int h) {
    #pragma unroll
    for (int q = 0; q < 8; ++q) {
        int kbB = ((q < 4) ? 2*q : 16 + 2*(q-4)) + 8*h;   // 16-bit B 32x16 VGPR layout
        B.u[q] = brow[(k + kbB) >> 1];
    }
}

__device__ __forceinline__ void async_ld_b128(unsigned lds_off, const char* gaddr) {
#if defined(__AMDGCN__)
    asm volatile("global_load_async_to_lds_b128 %0, %1, off"
                 :: "v"(lds_off), "v"(gaddr) : "memory");
#endif
}
__device__ __forceinline__ void wait_async0() {
#if defined(__AMDGCN__)
    asm volatile("s_wait_asynccnt 0x0" ::: "memory");
#endif
}

__global__ __launch_bounds__(256, 1)
void span_ngram_attn_kernel(const int*   __restrict__ ngram_seq,   // [C, L]
                            const float* __restrict__ span,        // [N, D]
                            const int*   __restrict__ maskm,       // [N, C, L]
                            const float* __restrict__ wemb,        // [NGRAM, D]
                            const float* __restrict__ cemb,        // [8, 1]
                            float*       __restrict__ out)         // [N, C*D]
{
    extern __shared__ char smem_raw[];
    _Float16* embS   = (_Float16*)smem_raw;                    // [256][PE]
    _Float16* embT   = embS + 256*PE;                          // [256][PE] transposed
    _Float16* aS     = embT + 256*PE;                          // [16][PA]
    _Float16* attS   = aS + 16*PA;                             // [16][PA]
    float*    rowsum = (float*)(attS + 16*PA);                 // [16]
    float*    stage  = rowsum + 16;                            // [16][256] f32 staging
    const unsigned stageOff = (unsigned)(uintptr_t)(const void*)stage;

    const int tid  = threadIdx.x;
    const int wave = tid >> 5;
    const int lane = tid & 31;
    const int m    = lane & 15;   // row (A) / col (B,C,D) within a 16-wide tile
    const int h    = lane >> 4;   // half-wave selector
    const int c    = blockIdx.y;
    const int nblock  = blockIdx.x * RPB;
    const int colbase = wave * 32;   // each wave owns 32 output columns

    // ---- kick off async DMA of span tile 0 (overlaps the emb gather) ----
    {
        const char* g = (const char*)(span + (size_t)nblock * DD);
        #pragma unroll
        for (int i = 0; i < 4; ++i) {
            int chunk = tid + i*256;             // 1024 x 16B = 16KB tile
            async_ld_b128(stageOff + chunk*16, g + (size_t)chunk*16);
        }
    }

    // ---- channel softmax weight (tiny, redundant per thread) ----
    float wch;
    {
        float s = 0.f, mine = 0.f;
        #pragma unroll
        for (int i = 0; i < CCH; ++i) {
            float e = __expf(cemb[i]);
            s += e;
            if (i == c) mine = e;
        }
        wch = mine / s;
    }

    // ---- gather channel embedding rows into LDS as f16 (row-major + transposed) ----
    for (int l = wave; l < LL; l += 8) {
        const int tok = ngram_seq[c*LL + l];
        const float* src = wemb + (size_t)tok * DD;
        #pragma unroll
        for (int j = 0; j < 8; ++j) {
            int d = j*32 + lane;                 // coalesced 128B per j
            _Float16 hv = (_Float16)src[d];
            embS[l*PE + d] = hv;                 // emb[l][d]
            embT[d*PE + l] = hv;                 // emb^T[d][l]
        }
    }
    __syncthreads();

    for (int st = 0; st < 4; ++st) {
        const int nbase = nblock + st*16;

        // ---- staged span tile ready once every wave waited + barrier ----
        wait_async0();
        __syncthreads();

        // convert staging f32 -> aS f16, folding 1/temper = 1/16
        #pragma unroll
        for (int i = 0; i < 16; ++i) {
            int idx = tid + i*256;
            int r = idx >> 8, col = idx & 255;
            aS[r*PA + col] = (_Float16)(stage[r*256 + col] * 0.0625f);
        }
        if (tid < 16) rowsum[tid] = 0.f;
        if (st < 3) { // warm L2 for next mask tile: global_prefetch_b8
            int rrow = tid >> 4, seg = tid & 15;
            int n = nbase + 16 + rrow;
            __builtin_prefetch((const char*)(maskm + ((size_t)n*CCH + c)*LL) + seg*64, 0, 1);
        }
        __syncthreads();

        // ---- DMA next span tile; overlaps everything below ----
        if (st < 3) {
            const char* g = (const char*)(span + (size_t)(nbase + 16) * DD);
            #pragma unroll
            for (int i = 0; i < 4; ++i) {
                int chunk = tid + i*256;
                async_ld_b128(stageOff + chunk*16, g + (size_t)chunk*16);
            }
        }

        // ---- GEMM1: u(16 x 32 l-cols) = spanTile(16xD) @ emb^T(DxL), K = D = 256 ----
        // Ping-pong fragment buffers so LDS loads overlap WMMA (partial s_wait_dscnt).
        v8f acc0 = {}; v8f acc1 = {};
        {
            const unsigned* arow  = (const unsigned*)(aS + m*PA);
            const unsigned* b0row = (const unsigned*)(embS + (size_t)(colbase + m)      * PE);
            const unsigned* b1row = (const unsigned*)(embS + (size_t)(colbase + 16 + m) * PE);
            FragU A[2], B0[2], B1[2];
            load_fragA(A[0], arow, 0, h);
            load_fragB(B0[0], b0row, 0, h);
            load_fragB(B1[0], b1row, 0, h);
            #pragma unroll
            for (int kt = 0; kt < 8; ++kt) {
                const int cur = kt & 1, nxt = cur ^ 1;
                if (kt < 7) {
                    load_fragA(A[nxt], arow, (kt+1)*32, h);
                    load_fragB(B0[nxt], b0row, (kt+1)*32, h);
                    load_fragB(B1[nxt], b1row, (kt+1)*32, h);
                }
                acc0 = __builtin_amdgcn_wmma_f32_16x16x32_f16(false, A[cur].v, false, B0[cur].v, (short)0, acc0, false, false);
                acc1 = __builtin_amdgcn_wmma_f32_16x16x32_f16(false, A[cur].v, false, B1[cur].v, (short)0, acc1, false, false);
            }
        }

        // ---- e = exp(u) * mask, stored UNNORMALIZED as f16 into attS ----
        #pragma unroll
        for (int t = 0; t < 2; ++t) {
            const int l = colbase + t*16 + m;        // C/D layout: lane = column
            const v8f acc = t ? acc1 : acc0;
            #pragma unroll
            for (int q = 0; q < 8; ++q) {
                int r = q + 8*h;                     // C/D layout: VGPR q -> row q+8h
                int n = nbase + r;
                int mk = maskm[((size_t)n*CCH + c)*LL + l];
                float e = (mk != 0) ? __expf(acc[q]) : 0.f;
                attS[r*PA + l] = (_Float16)e;
            }
        }
        __syncthreads();

        // ---- row sums via the matrix pipe: partial = e_slice @ ones(32x16) ----
        // Wave w reduces K-slice kt = w with a single WMMA; column 0 lanes own the sums.
        {
            FragU A, Ones;
            #pragma unroll
            for (int q = 0; q < 8; ++q) Ones.u[q] = 0x3C003C00u;   // packed f16 1.0
            const unsigned* arow = (const unsigned*)(attS + m*PA);
            load_fragA(A, arow, wave*32, h);
            v8f ps = {};
            ps = __builtin_amdgcn_wmma_f32_16x16x32_f16(false, A.v, false, Ones.v, (short)0, ps, false, false);
            if (m == 0) {                            // lanes 0 (rows 0-7) and 16 (rows 8-15)
                #pragma unroll
                for (int q = 0; q < 8; ++q)
                    atomicAdd(rowsum + (q + 8*h), ps[q]);   // ds_add_f32
            }
        }

        // ---- GEMM2: outTile(16 x 32 d-cols) = e(16xL) @ emb(LxD), K = L = 256 ----
        // Normalization is folded into the store (diag(1/rowsum) on the left).
        v8f o0 = {}; v8f o1 = {};
        {
            const unsigned* arow  = (const unsigned*)(attS + m*PA);
            const unsigned* b0row = (const unsigned*)(embT + (size_t)(colbase + m)      * PE);
            const unsigned* b1row = (const unsigned*)(embT + (size_t)(colbase + 16 + m) * PE);
            FragU A[2], B0[2], B1[2];
            load_fragA(A[0], arow, 0, h);
            load_fragB(B0[0], b0row, 0, h);
            load_fragB(B1[0], b1row, 0, h);
            #pragma unroll
            for (int kt = 0; kt < 8; ++kt) {
                const int cur = kt & 1, nxt = cur ^ 1;
                if (kt < 7) {
                    load_fragA(A[nxt], arow, (kt+1)*32, h);
                    load_fragB(B0[nxt], b0row, (kt+1)*32, h);
                    load_fragB(B1[nxt], b1row, (kt+1)*32, h);
                }
                o0 = __builtin_amdgcn_wmma_f32_16x16x32_f16(false, A[cur].v, false, B0[cur].v, (short)0, o0, false, false);
                o1 = __builtin_amdgcn_wmma_f32_16x16x32_f16(false, A[cur].v, false, B1[cur].v, (short)0, o1, false, false);
            }
        }
        __syncthreads();   // rowsum atomics complete; attS/aS free for next tile

        // ---- scale by wch / (rowsum + 1e-10) and store: out[n][c*D + d] ----
        float scl[8];
        #pragma unroll
        for (int q = 0; q < 8; ++q)
            scl[q] = wch / (rowsum[q + 8*h] + 1e-10f);
        #pragma unroll
        for (int t = 0; t < 2; ++t) {
            const int d = colbase + t*16 + m;
            const v8f acc = t ? o1 : o0;
            #pragma unroll
            for (int q = 0; q < 8; ++q) {
                int r = q + 8*h;
                int n = nbase + r;
                out[(size_t)n*CD + c*DD + d] = acc[q] * scl[q];
            }
        }
        // next iteration's wait+barrier fences LDS reuse
    }
}

extern "C" void kernel_launch(void* const* d_in, const int* in_sizes, int n_in,
                              void* d_out, int out_size, void* d_ws, size_t ws_size,
                              hipStream_t stream) {
    (void)in_sizes; (void)n_in; (void)out_size; (void)d_ws; (void)ws_size;
    const int*   ngram = (const int*)  d_in[0];   // ngram_seq [8,256]
    const float* span  = (const float*)d_in[1];   // span_matrix [128,128,256]
    const int*   maskm = (const int*)  d_in[2];   // span_ngram_mask_matrix [128,128,8,256]
    /* d_in[3] = channel_ids (arange, unused) */
    const float* wemb  = (const float*)d_in[4];   // word_emb [50000,256]
    const float* cemb  = (const float*)d_in[5];   // channel_emb [8,1]
    float* out = (float*)d_out;                   // [128,128,2048]

    hipFuncSetAttribute(reinterpret_cast<const void*>(span_ngram_attn_kernel),
                        hipFuncAttributeMaxDynamicSharedMemorySize, SMEM_BYTES);

    dim3 grid(NSP / RPB, CCH);   // 256 x 8 blocks
    dim3 block(256);             // 8 wave32
    span_ngram_attn_kernel<<<grid, block, SMEM_BYTES, stream>>>(
        ngram, span, maskm, wemb, cemb, out);
}
